// GATEncoder_14336600834091
// MI455X (gfx1250) — matrix operand order repaired and checked by reading, not compiled, not run on hardware
//
#include <hip/hip_runtime.h>
#include <hip/hip_bf16.h>

typedef __attribute__((ext_vector_type(16))) __bf16 v16bf;
typedef __attribute__((ext_vector_type(8)))  __bf16 v8bf;
typedef __attribute__((ext_vector_type(8)))  float  v8f;

#define NEG_SLOPE 0.2f

__device__ __forceinline__ void atomicMaxFloat(float* addr, float val) {
    // mixed-sign safe float max via int max / uint min
    if (val >= 0.0f) atomicMax((int*)addr, __float_as_int(val));
    else             atomicMin((unsigned int*)addr, __float_as_uint(val));
}

// ---------------- prep ----------------
__global__ void k_f32_to_bf16(const float* __restrict__ in, __bf16* __restrict__ out, int n) {
    int i = blockIdx.x * blockDim.x + threadIdx.x;
    if (i < n) out[i] = (__bf16)in[i];
}

// W row-major [K,Nout] -> Wt row-major [Nout,K] in bf16
__global__ void k_transpose_bf16(const float* __restrict__ W, __bf16* __restrict__ Wt, int K, int Nout) {
    int i = blockIdx.x * blockDim.x + threadIdx.x;
    if (i >= K * Nout) return;
    int k = i / Nout, n = i - k * Nout;
    Wt[(size_t)n * K + k] = (__bf16)W[i];
}

__global__ void k_init(float* __restrict__ accum, float* __restrict__ emax,
                       float* __restrict__ denom, int N) {
    int i = blockIdx.x * blockDim.x + threadIdx.x;
    if (i < N * 128) accum[i] = 0.0f;
    if (i < N * 2) { emax[i] = -__builtin_inff(); denom[i] = 0.0f; }
}

// ---------------- dual WMMA GEMM: C1 = A*B1t^T, C2 = A*B2t^T ----------------
// A: [N,K] bf16 row-major. B1t/B2t: [128,K] bf16 row-major (pre-transposed weights).
// C1/C2: [N,128] f32 row-major. One wave per 16x16 tile; 8 waves cover Nout=128.
// A fragment (16x32, contiguous k order, not the ISA table order): lane-half h holds 16 consecutive K starting at h*16.
// B fragment (32x16, same contiguous k order as A): lane-half h holds 16 consecutive K starting at h*16.
__global__ __launch_bounds__(256)
void k_gemm_dual(const __bf16* __restrict__ A, const __bf16* __restrict__ B1t,
                 const __bf16* __restrict__ B2t, float* __restrict__ C1,
                 float* __restrict__ C2, int N, int K) {
    const int lane = threadIdx.x & 31;
    const int wave = threadIdx.x >> 5;
    const int m0   = blockIdx.x * 16;
    const int n0   = wave * 16;
    const int r    = lane & 15;
    const int h    = lane >> 4;

    int arow = m0 + r; if (arow >= N) arow = N - 1;
    const __bf16* ap  = A   + (size_t)arow * K     + h * 16;  // one contiguous 16-elem chunk
    const __bf16* b1p = B1t + (size_t)(n0 + r) * K + h * 16;  // one contiguous 16-elem chunk
    const __bf16* b2p = B2t + (size_t)(n0 + r) * K + h * 16;

    v8f acc1 = {}; v8f acc2 = {};
    for (int k0 = 0; k0 < K; k0 += 32) {
        v16bf a  = *(const v16bf*)(ap + k0);
        v16bf b1 = *(const v16bf*)(b1p + k0);
        v16bf b2 = *(const v16bf*)(b2p + k0);
        acc1 = __builtin_amdgcn_wmma_f32_16x16x32_bf16(false, a, false, b1, (short)0, acc1, false, false);
        acc2 = __builtin_amdgcn_wmma_f32_16x16x32_bf16(false, a, false, b2, (short)0, acc2, false, false);
    }
    #pragma unroll
    for (int i = 0; i < 8; ++i) {
        int row = m0 + h * 8 + i;
        if (row < N) {
            C1[(size_t)row * 128 + n0 + r] = acc1[i];
            C2[(size_t)row * 128 + n0 + r] = acc2[i];
        }
    }
}

// ---------------- edge pass 1: logits + segment max ----------------
// one wave per edge; lane l covers features [4l..4l+3]; lanes 0-15 = head 0, 16-31 = head 1
__global__ void k_edge_logits(const float* __restrict__ XL, const float* __restrict__ XR,
                              const float* __restrict__ att, const int* __restrict__ src,
                              const int* __restrict__ tgt, int E, int ET,
                              float* __restrict__ elog, float* __restrict__ emax) {
    int wid  = (blockIdx.x * blockDim.x + threadIdx.x) >> 5;
    int lane = threadIdx.x & 31;
    if (wid >= ET) return;
    int s, t;
    if (wid < E) { s = src[wid]; t = tgt[wid]; } else { s = t = wid - E; }
    float4 xl4 = ((const float4*)(XL + (size_t)s * 128))[lane];
    float4 xr4 = ((const float4*)(XR + (size_t)t * 128))[lane];
    float4 at4 = ((const float4*)att)[lane];
    float m, sum = 0.0f;
    m = xl4.x + xr4.x; m = (m > 0.0f) ? m : NEG_SLOPE * m; sum += m * at4.x;
    m = xl4.y + xr4.y; m = (m > 0.0f) ? m : NEG_SLOPE * m; sum += m * at4.y;
    m = xl4.z + xr4.z; m = (m > 0.0f) ? m : NEG_SLOPE * m; sum += m * at4.z;
    m = xl4.w + xr4.w; m = (m > 0.0f) ? m : NEG_SLOPE * m; sum += m * at4.w;
    // reduce within each 16-lane half (per head)
    for (int off = 1; off < 16; off <<= 1) sum += __shfl_xor(sum, off, 32);
    int head = lane >> 4;
    if ((lane & 15) == 0) {
        elog[(size_t)wid * 2 + head] = sum;
        atomicMaxFloat(&emax[(size_t)t * 2 + head], sum);
    }
}

// ---------------- edge pass 2 (fused): alpha = exp(e-max), denom += alpha, scatter ----------------
__global__ void k_alpha_scatter(const float* __restrict__ XL, const float* __restrict__ elog,
                                const float* __restrict__ emax, const int* __restrict__ src,
                                const int* __restrict__ tgt, int E, int ET,
                                float* __restrict__ denom, float* __restrict__ accum) {
    int wid  = (blockIdx.x * blockDim.x + threadIdx.x) >> 5;
    int lane = threadIdx.x & 31;
    if (wid >= ET) return;
    int s, t;
    if (wid < E) { s = src[wid]; t = tgt[wid]; } else { s = t = wid - E; }
    int head = lane >> 4;
    float a = __expf(elog[(size_t)wid * 2 + head] - emax[(size_t)t * 2 + head]);
    if ((lane & 15) == 0) atomicAdd(&denom[(size_t)t * 2 + head], a);
    float4 xl4 = ((const float4*)(XL + (size_t)s * 128))[lane];
    float* dst = accum + (size_t)t * 128 + lane * 4;
    atomicAdd(dst + 0, a * xl4.x);
    atomicAdd(dst + 1, a * xl4.y);
    atomicAdd(dst + 2, a * xl4.z);
    atomicAdd(dst + 3, a * xl4.w);
}

// ---------------- node finalize ----------------
__global__ void k_finalize1(const float* __restrict__ accum, const float* __restrict__ denom,
                            const float* __restrict__ bias, __bf16* __restrict__ out, int N) {
    int i = blockIdx.x * blockDim.x + threadIdx.x;
    if (i >= N * 64) return;
    int n = i >> 6, c = i & 63;
    float d0 = denom[(size_t)n * 2 + 0] + 1e-16f;
    float d1 = denom[(size_t)n * 2 + 1] + 1e-16f;
    float v = 0.5f * (accum[(size_t)n * 128 + c] / d0 + accum[(size_t)n * 128 + 64 + c] / d1) + bias[c];
    out[i] = (__bf16)fmaxf(v, 0.0f);   // ReLU between layers, bf16 for next GEMM
}

__global__ void k_finalize2(const float* __restrict__ accum, const float* __restrict__ denom,
                            const float* __restrict__ bias, float* __restrict__ out, int N) {
    int i = blockIdx.x * blockDim.x + threadIdx.x;
    if (i >= N * 64) return;
    int n = i >> 6, c = i & 63;
    float d0 = denom[(size_t)n * 2 + 0] + 1e-16f;
    float d1 = denom[(size_t)n * 2 + 1] + 1e-16f;
    out[i] = 0.5f * (accum[(size_t)n * 128 + c] / d0 + accum[(size_t)n * 128 + 64 + c] / d1) + bias[c];
}

extern "C" void kernel_launch(void* const* d_in, const int* in_sizes, int n_in,
                              void* d_out, int out_size, void* d_ws, size_t ws_size,
                              hipStream_t stream) {
    const float* features = (const float*)d_in[0];
    const int*   edges    = (const int*)d_in[1];
    const float* Wl1  = (const float*)d_in[2];
    const float* Wr1  = (const float*)d_in[3];
    const float* att1 = (const float*)d_in[4];
    const float* b1   = (const float*)d_in[5];
    const float* Wl2  = (const float*)d_in[6];
    const float* Wr2  = (const float*)d_in[7];
    const float* att2 = (const float*)d_in[8];
    const float* b2   = (const float*)d_in[9];

    const int N  = in_sizes[0] / 128;
    const int E  = in_sizes[1] / 2;
    const int ET = E + N;
    const int* src = edges;
    const int* tgt = edges + E;

    char* ws = (char*)d_ws;
    size_t off = 0;
    auto alloc = [&](size_t bytes) { void* p = ws + off; off = (off + bytes + 255) & ~(size_t)255; return p; };

    __bf16* xbf   = (__bf16*)alloc((size_t)N * 128 * 2);
    __bf16* wl1t  = (__bf16*)alloc((size_t)128 * 128 * 2);
    __bf16* wr1t  = (__bf16*)alloc((size_t)128 * 128 * 2);
    __bf16* wl2t  = (__bf16*)alloc((size_t)128 * 64 * 2);
    __bf16* wr2t  = (__bf16*)alloc((size_t)128 * 64 * 2);
    float*  XL    = (float*)alloc((size_t)N * 128 * 4);
    float*  XR    = (float*)alloc((size_t)N * 128 * 4);
    float*  elog  = (float*)alloc((size_t)ET * 2 * 4);
    float*  emax  = (float*)alloc((size_t)N * 2 * 4);
    float*  denom = (float*)alloc((size_t)N * 2 * 4);
    float*  accum = (float*)alloc((size_t)N * 128 * 4);
    __bf16* h1bf  = (__bf16*)alloc((size_t)N * 64 * 2);

    const int T = 256;
    // prep: bf16 conversions + transposed weights
    k_f32_to_bf16<<<(N * 128 + T - 1) / T, T, 0, stream>>>(features, xbf, N * 128);
    k_transpose_bf16<<<(128 * 128 + T - 1) / T, T, 0, stream>>>(Wl1, wl1t, 128, 128);
    k_transpose_bf16<<<(128 * 128 + T - 1) / T, T, 0, stream>>>(Wr1, wr1t, 128, 128);
    k_transpose_bf16<<<(64 * 128 + T - 1) / T, T, 0, stream>>>(Wl2, wl2t, 64, 128);
    k_transpose_bf16<<<(64 * 128 + T - 1) / T, T, 0, stream>>>(Wr2, wr2t, 64, 128);

    // ---- layer 1 ----
    k_init<<<(N * 128 + T - 1) / T, T, 0, stream>>>(accum, emax, denom, N);
    k_gemm_dual<<<(N + 15) / 16, T, 0, stream>>>(xbf, wl1t, wr1t, XL, XR, N, 128);
    k_edge_logits<<<(ET + 7) / 8, T, 0, stream>>>(XL, XR, att1, src, tgt, E, ET, elog, emax);
    k_alpha_scatter<<<(ET + 7) / 8, T, 0, stream>>>(XL, elog, emax, src, tgt, E, ET, denom, accum);
    k_finalize1<<<(N * 64 + T - 1) / T, T, 0, stream>>>(accum, denom, b1, h1bf, N);

    // ---- layer 2 ----
    k_init<<<(N * 128 + T - 1) / T, T, 0, stream>>>(accum, emax, denom, N);
    k_gemm_dual<<<(N + 15) / 16, T, 0, stream>>>(h1bf, wl2t, wr2t, XL, XR, N, 64);
    k_edge_logits<<<(ET + 7) / 8, T, 0, stream>>>(XL, XR, att2, src, tgt, E, ET, elog, emax);
    k_alpha_scatter<<<(ET + 7) / 8, T, 0, stream>>>(XL, elog, emax, src, tgt, E, ET, denom, accum);
    k_finalize2<<<(N * 64 + T - 1) / T, T, 0, stream>>>(accum, denom, b2, (float*)d_out, N);
}
